// GATDiscriminator_59425167508073
// MI455X (gfx1250) — matrix operand order repaired and checked by reading, not compiled, not run on hardware
//
#include <hip/hip_runtime.h>
#include <hip/hip_bf16.h>
#include <math.h>

#define HEADS 8
#define OUTD 16
#define FDIM 128            // HEADS*OUTD == input dim == hidden dim
#define NEG_SLOPE 0.2f

typedef __attribute__((ext_vector_type(16))) _Float16 v16h;
typedef __attribute__((ext_vector_type(8)))  _Float16 v8h;
typedef __attribute__((ext_vector_type(8)))  float    v8f;

// ---- order-preserving float<->uint encoding for atomicMax-based segment max ----
static __device__ __forceinline__ unsigned fenc(float f) {
    unsigned u = __float_as_uint(f);
    return (u & 0x80000000u) ? ~u : (u | 0x80000000u);
}
static __device__ __forceinline__ float fdec(unsigned u) {
    unsigned b = (u & 0x80000000u) ? (u & 0x7fffffffu) : ~u;
    return __uint_as_float(b);
}
#define ENC_NEG_INF 0x007FFFFFu   // fenc(-inf)

// ---------------- f32 -> f16 convert (GEMM input) ----------------
__global__ __launch_bounds__(256) void k_cvt_f16(const float* __restrict__ x,
                                                 _Float16* __restrict__ xh, int n) {
    int i = blockIdx.x * blockDim.x + threadIdx.x;
    if (i < n) xh[i] = (_Float16)x[i];
}

// ---------------- pack W[k][n] (128x128 row-major f32) into WMMA B-fragment layout ----
// fragment (kt,nt): lane l, elem e -> K = kt*32 + (e<8 ? 8*(l>>4)+e : 16+8*(l>>4)+(e-8)), N = nt*16 + (l&15)
__global__ __launch_bounds__(256) void k_pack_w(const float* __restrict__ W,
                                                _Float16* __restrict__ Wp) {
    int i = blockIdx.x * blockDim.x + threadIdx.x;   // [0, 4*8*32*16)
    if (i >= 4 * 8 * 32 * 16) return;
    int e  = i & 15;
    int l  = (i >> 4) & 31;
    int nt = (i >> 9) & 7;
    int kt = (i >> 12);
    int K  = kt * 32 + ((e < 8) ? 8 * (l >> 4) + e : 16 + 8 * (l >> 4) + (e - 8));
    int Nc = nt * 16 + (l & 15);
    Wp[i] = (_Float16)W[K * FDIM + Nc];
}

// ---------------- H = Xh @ W via v_wmma_f32_16x16x32_f16, one wave per 16x16 tile ----
__global__ __launch_bounds__(256) void k_gemm_wmma(const _Float16* __restrict__ Xh,
                                                   const _Float16* __restrict__ Wp,
                                                   float* __restrict__ H, int n) {
    int lane   = threadIdx.x & 31;
    int wave   = (blockIdx.x * blockDim.x + threadIdx.x) >> 5;
    int nWaves = (gridDim.x * blockDim.x) >> 5;
    int mTiles = (n + 15) >> 4;
    int tiles  = mTiles * 8;                 // 8 N-tiles of 16 (FDIM=128)
    for (int t = wave; t < tiles; t += nWaves) {
        int mt = t >> 3, nt = t & 7;
        int m0 = mt << 4;
        bool full = (m0 + 16 <= n);          // wave-uniform: full tile, no guards
        int row = m0 + (lane & 15);
        if (!full && row >= n) row = n - 1;  // clamp loads only on the ragged tile
        v8f c = {};
        #pragma unroll
        for (int kt = 0; kt < 4; ++kt) {     // K = 128 = 4 x 32
            // A-fragment (16x32 f16): lane<16 holds K[kk..kk+7],[kk+16..kk+23];
            //                          lane>=16 holds K[kk+8..],[kk+24..]
            int kA = kt * 32 + ((lane < 16) ? 0 : 8);
            v8h lo = *(const v8h*)(Xh + row * FDIM + kA);
            v8h hi = *(const v8h*)(Xh + row * FDIM + kA + 16);
            v16h a;
            #pragma unroll
            for (int i = 0; i < 8; ++i) { a[i] = lo[i]; a[8 + i] = hi[i]; }
            // B-fragment: contiguous 16 halves per lane from the packed buffer
            v16h b = *(const v16h*)(Wp + ((kt * 8 + nt) * 32 + lane) * 16);
            c = __builtin_amdgcn_wmma_f32_16x16x32_f16(
                    false, a, false, b, (short)0, c, false, false);
        }
        // C/D layout: vgpr r -> M = (lane<16 ? r : 8+r), N = lane&15
        int col   = nt * 16 + (lane & 15);
        int rbase = m0 + ((lane < 16) ? 0 : 8);
        float* hp = H + rbase * FDIM + col;
        if (full) {                          // uniform branch: straight-line stores
            #pragma unroll
            for (int r = 0; r < 8; ++r) hp[r * FDIM] = c[r];
        } else {
            #pragma unroll
            for (int r = 0; r < 8; ++r) {
                if (rbase + r < n) hp[r * FDIM] = c[r];
            }
        }
    }
}

// ---------------- per-node attention logits: alpha_s/d[n,h] = <h[n,h,:], a_src/dst[h,:]> ----
__global__ __launch_bounds__(256) void k_alpha(const float* __restrict__ H,
                                               const float* __restrict__ a_src,
                                               const float* __restrict__ a_dst,
                                               float* __restrict__ as_,
                                               float* __restrict__ ad_, int n) {
    int i = blockIdx.x * blockDim.x + threadIdx.x;
    if (i >= n * HEADS) return;
    int node = i >> 3, hh = i & 7;
    const float* hp = H + node * FDIM + hh * OUTD;
    const float* s  = a_src + hh * OUTD;
    const float* d  = a_dst + hh * OUTD;
    float ss = 0.f, dd = 0.f;
    #pragma unroll
    for (int o = 0; o < OUTD; ++o) { float hv = hp[o]; ss += hv * s[o]; dd += hv * d[o]; }
    as_[i] = ss; ad_[i] = dd;
}

// ---------------- init reduction buffers ----------------
__global__ __launch_bounds__(256) void k_init(unsigned* __restrict__ emax,
                                              float* __restrict__ denom,
                                              float* __restrict__ accum, int n) {
    int i = blockIdx.x * blockDim.x + threadIdx.x;
    if (i < n * FDIM) accum[i] = 0.f;
    if (i < n * HEADS) { emax[i] = ENC_NEG_INF; denom[i] = 0.f; }
}

// ---------------- pass 1 over edges: segment max of leaky-relu logits ----------------
__global__ __launch_bounds__(256) void k_edge_max(const int* __restrict__ ei, int E, int n,
                                                  const float* __restrict__ as_,
                                                  const float* __restrict__ ad_,
                                                  unsigned* __restrict__ emax) {
    int i = blockIdx.x * blockDim.x + threadIdx.x;
    int tot = (E + n) * HEADS;
    if (i >= tot) return;
    int e = i >> 3, hh = i & 7;
    int s, d;
    if (e < E) { s = ei[e]; d = ei[E + e]; } else { s = d = e - E; }   // self loops
    float v = as_[s * HEADS + hh] + ad_[d * HEADS + hh];
    v = v > 0.f ? v : NEG_SLOPE * v;
    atomicMax(emax + d * HEADS + hh, fenc(v));
}

// ---------------- pass 2 over edges (fused): a = exp(e - max); denom += a; accum += a*h[src]
// one wave per edge: 32 lanes x 4 channels = 128 feature channels; head = lane>>2
__global__ __launch_bounds__(256) void k_edge_agg(const int* __restrict__ ei, int E, int n,
                                                  const float* __restrict__ as_,
                                                  const float* __restrict__ ad_,
                                                  const unsigned* __restrict__ emax,
                                                  const float* __restrict__ H,
                                                  float* __restrict__ denom,
                                                  float* __restrict__ accum) {
    int lane   = threadIdx.x & 31;
    int wave   = (blockIdx.x * blockDim.x + threadIdx.x) >> 5;
    int nWaves = (gridDim.x * blockDim.x) >> 5;
    int tot    = E + n;
    int hh = lane >> 2;
    int c0 = lane << 2;
    for (int e = wave; e < tot; e += nWaves) {
        int s, d;
        if (e < E) { s = ei[e]; d = ei[E + e]; } else { s = d = e - E; }
        float v = as_[s * HEADS + hh] + ad_[d * HEADS + hh];
        v = v > 0.f ? v : NEG_SLOPE * v;
        float a = __expf(v - fdec(emax[d * HEADS + hh]));
        if ((lane & 3) == 0) atomicAdd(denom + d * HEADS + hh, a);
        const float4 hv = *(const float4*)(H + s * FDIM + c0);
        float* acc = accum + d * FDIM + c0;
        atomicAdd(acc + 0, hv.x * a);
        atomicAdd(acc + 1, hv.y * a);
        atomicAdd(acc + 2, hv.z * a);
        atomicAdd(acc + 3, hv.w * a);
    }
}

// ---------------- normalize, bias, tanh; emit f16 (next GEMM) and/or f32 ----------------
__global__ __launch_bounds__(256) void k_finalize(const float* __restrict__ accum,
                                                  const float* __restrict__ denom,
                                                  const float* __restrict__ bias,
                                                  _Float16* __restrict__ xh,
                                                  float* __restrict__ xf, int n) {
    int i = blockIdx.x * blockDim.x + threadIdx.x;
    if (i >= n * FDIM) return;
    int node = i >> 7, c = i & 127, hh = c >> 4;
    float t = tanhf(accum[i] / denom[node * HEADS + hh] + bias[c]);
    if (xh) xh[i] = (_Float16)t;
    if (xf) xf[i] = t;
}

// ---------------- final 128->1 linear head; one wave per node ----------------
__global__ __launch_bounds__(256) void k_linear(const float* __restrict__ X,
                                                const float* __restrict__ lw,
                                                const float* __restrict__ lb,
                                                float* __restrict__ out, int n) {
    int lane   = threadIdx.x & 31;
    int wave   = (blockIdx.x * blockDim.x + threadIdx.x) >> 5;
    int nWaves = (gridDim.x * blockDim.x) >> 5;
    for (int node = wave; node < n; node += nWaves) {
        const float4 xv = *(const float4*)(X + node * FDIM + lane * 4);
        const float4 wv = *(const float4*)(lw + lane * 4);
        float sum = xv.x * wv.x + xv.y * wv.y + xv.z * wv.z + xv.w * wv.w;
        #pragma unroll
        for (int off = 16; off > 0; off >>= 1) sum += __shfl_xor(sum, off, 32);
        if (lane == 0) out[node] = sum + lb[0];
    }
}

extern "C" void kernel_launch(void* const* d_in, const int* in_sizes, int n_in,
                              void* d_out, int out_size, void* d_ws, size_t ws_size,
                              hipStream_t stream) {
    const float* z   = (const float*)d_in[0];
    const int*   ei  = (const int*)d_in[1];
    const float* W1  = (const float*)d_in[2];
    const float* a1s = (const float*)d_in[3];
    const float* a1d = (const float*)d_in[4];
    const float* b1  = (const float*)d_in[5];
    const float* W2  = (const float*)d_in[6];
    const float* a2s = (const float*)d_in[7];
    const float* a2d = (const float*)d_in[8];
    const float* b2  = (const float*)d_in[9];
    const float* lw  = (const float*)d_in[10];
    const float* lb  = (const float*)d_in[11];
    const int n = in_sizes[0] / FDIM;   // 50000 nodes
    const int E = in_sizes[1] / 2;      // 1,600,000 edges

    // workspace carve-out (~71 MB); node-level arrays stay L2-resident on MI455X
    char* ws = (char*)d_ws;
    size_t off = 0;
    auto alloc = [&](size_t bytes) -> void* {
        void* p = ws + off;
        off += (bytes + 255) & ~(size_t)255;
        return p;
    };
    _Float16* Xh   = (_Float16*)alloc((size_t)n * FDIM * 2);        // f16 GEMM input
    _Float16* Wp   = (_Float16*)alloc((size_t)4 * 8 * 32 * 16 * 2); // packed B fragments
    float*    H    = (float*)   alloc((size_t)n * FDIM * 4);        // h = x@W (f32)
    float*    as_  = (float*)   alloc((size_t)n * HEADS * 4);
    float*    ad_  = (float*)   alloc((size_t)n * HEADS * 4);
    unsigned* emax = (unsigned*)alloc((size_t)n * HEADS * 4);
    float*    den  = (float*)   alloc((size_t)n * HEADS * 4);
    float*    acc  = (float*)   alloc((size_t)n * FDIM * 4);
    (void)ws_size; (void)n_in; (void)out_size;

    const int T = 256;
    const int gNF   = (n * FDIM + T - 1) / T;
    const int gNH   = (n * HEADS + T - 1) / T;
    const int gEM   = ((E + n) * HEADS + T - 1) / T;
    const int gAGG  = ((E + n) * 32 + T - 1) / T;                 // one wave32 per edge
    const int gGEMM = (((n + 15) / 16) * 8 * 32 + T - 1) / T;     // one wave per 16x16 tile

    auto run_layer = [&](const float* W, const float* asw, const float* adw,
                         const float* bw, _Float16* xh_out, float* xf_out) {
        k_pack_w   <<<64,    T, 0, stream>>>(W, Wp);
        k_gemm_wmma<<<gGEMM, T, 0, stream>>>(Xh, Wp, H, n);
        k_alpha    <<<gNH,   T, 0, stream>>>(H, asw, adw, as_, ad_, n);
        k_init     <<<gNF,   T, 0, stream>>>(emax, den, acc, n);
        k_edge_max <<<gEM,   T, 0, stream>>>(ei, E, n, as_, ad_, emax);
        k_edge_agg <<<gAGG,  T, 0, stream>>>(ei, E, n, as_, ad_, emax, H, den, acc);
        k_finalize <<<gNF,   T, 0, stream>>>(acc, den, bw, xh_out, xf_out, n);
    };

    k_cvt_f16<<<gNF, T, 0, stream>>>(z, Xh, n * FDIM);
    run_layer(W1, a1s, a1d, b1, Xh, nullptr);   // layer 1 -> f16 input for layer 2
    run_layer(W2, a2s, a2d, b2, nullptr, H);    // layer 2 -> f32 x2 (reuse H, no longer read)
    k_linear<<<(n * 32 + T - 1) / T, T, 0, stream>>>(H, lw, lb, (float*)d_out, n);
}
